// LinearMem_7533372638057
// MI455X (gfx1250) — matrix operand order repaired and checked by reading, not compiled, not run on hardware
//
#include <hip/hip_runtime.h>
#include <hip/hip_bf16.h>

typedef __attribute__((ext_vector_type(16))) _Float16 v16h;
typedef __attribute__((ext_vector_type(8)))  _Float16 v8h;
typedef __attribute__((ext_vector_type(4)))  _Float16 v4h;
typedef __attribute__((ext_vector_type(8)))  float    v8f;

// 16-byte payload type matching the async-LDS builtin's parameter type
typedef int lm_b128 __attribute__((vector_size(16)));

// ---- CDNA5 async global->LDS path (guarded; falls back to sync copy) ------
#if __has_builtin(__builtin_amdgcn_global_load_async_to_lds_b128)
#define LM_HAS_ASYNC 1
#else
#define LM_HAS_ASYNC 0
#endif

#if __has_builtin(__builtin_amdgcn_s_wait_asynccnt)
#define LM_WAIT_ASYNC(n) __builtin_amdgcn_s_wait_asynccnt(n)
#elif LM_HAS_ASYNC
#define LM_WAIT_ASYNC(n) asm volatile("s_wait_asynccnt %0" ::"i"(n) : "memory")
#else
#define LM_WAIT_ASYNC(n)
#endif

__device__ __forceinline__ void lm_copy16(const _Float16* g, _Float16* l) {
#if LM_HAS_ASYNC
    __builtin_amdgcn_global_load_async_to_lds_b128(
        (__attribute__((address_space(1))) lm_b128*)g,
        (__attribute__((address_space(3))) lm_b128*)l,
        /*offset=*/0, /*cpol=*/0);
#else
    *(int4*)l = *(const int4*)g;
#endif
}

// ---------------------------------------------------------------------------
// Kernel 1: quantize x with granularity (1,32). One wave per 32-element block.
// Writes q values as f16 (exact integers) and inv-scale transposed [K/32][T].
// ---------------------------------------------------------------------------
__global__ __launch_bounds__(256) void lm_quant_x(const float* __restrict__ x,
                                                  _Float16* __restrict__ xq,
                                                  float* __restrict__ sxT,
                                                  int T, int K) {
    const int lane = threadIdx.x & 31;
    const int gw   = blockIdx.x * 8 + (threadIdx.x >> 5);   // global wave id
    const int KB   = K >> 5;
    const int t    = gw / KB;
    const int b    = gw % KB;
    const size_t idx = (size_t)t * K + (size_t)b * 32 + lane;

    float v = x[idx];
    float m = fabsf(v);
#pragma unroll
    for (int off = 16; off > 0; off >>= 1)
        m = fmaxf(m, __shfl_xor(m, off, 32));
    if (m == 0.0f) m = 1.0f;
    const float scale = 127.0f / m;
    float q = fminf(fmaxf(rintf(v * scale), -127.0f), 127.0f);  // RNE == jnp.round
    xq[idx] = (_Float16)q;
    if (lane == 0) sxT[(size_t)b * T + t] = m / 127.0f;   // dequant factor
}

// ---------------------------------------------------------------------------
// Kernel 2: quantize weight with 32x32 blocks (blocks of weight.T == same
// blocks of weight [N,K]). One 256-thread workgroup per block; wq kept in
// [N,K] layout (exactly the B-fragment layout the WMMA loader wants).
// ---------------------------------------------------------------------------
__global__ __launch_bounds__(256) void lm_quant_w(const float* __restrict__ w,
                                                  _Float16* __restrict__ wq,
                                                  float* __restrict__ sw,
                                                  int N, int K) {
    __shared__ float red[8];
    const int NB  = N >> 5;
    const int nb  = blockIdx.x % NB;   // block along N
    const int kb  = blockIdx.x / NB;   // block along K
    const int tid = threadIdx.x;
    const int n   = nb * 32 + (tid >> 3);
    const int k   = kb * 32 + (tid & 7) * 4;
    const size_t idx = (size_t)n * K + k;

    const float4 v = *(const float4*)&w[idx];
    float m = fmaxf(fmaxf(fabsf(v.x), fabsf(v.y)), fmaxf(fabsf(v.z), fabsf(v.w)));
#pragma unroll
    for (int off = 16; off > 0; off >>= 1)
        m = fmaxf(m, __shfl_xor(m, off, 32));
    if ((tid & 31) == 0) red[tid >> 5] = m;
    __syncthreads();
    m = red[0];
#pragma unroll
    for (int i = 1; i < 8; ++i) m = fmaxf(m, red[i]);
    if (m == 0.0f) m = 1.0f;
    const float scale = 127.0f / m;

    v4h q;
    q.x = (_Float16)fminf(fmaxf(rintf(v.x * scale), -127.0f), 127.0f);
    q.y = (_Float16)fminf(fmaxf(rintf(v.y * scale), -127.0f), 127.0f);
    q.z = (_Float16)fminf(fmaxf(rintf(v.z * scale), -127.0f), 127.0f);
    q.w = (_Float16)fminf(fmaxf(rintf(v.w * scale), -127.0f), 127.0f);
    *(v4h*)&wq[idx] = q;
    if (tid == 0) sw[(size_t)kb * NB + nb] = m / 127.0f;
}

// ---------------------------------------------------------------------------
// Kernel 3: WMMA GEMM, async double-buffered LDS staging.
// 256 threads (8 waves) per 128x128 output tile; waves 4(M) x 2(N), each wave
// owns 2x4 WMMA tiles. Per 32-wide k-block: GLOBAL_LOAD_ASYNC_TO_LDS_B128
// stages A/B tiles (1-deep prefetch, ASYNCcnt-managed), then 8x
// v_wmma_f32_16x16x32_f16 with C=0, software-pipelined against the f32
// scale-FMA (per-row input scale via v8f load from transposed sxT, per-32-N
// weight scale as a wave scalar).
// ---------------------------------------------------------------------------
#define LDP 40   // padded row length in halves (16B aligned, conflict-free)

__global__ __launch_bounds__(256) void lm_wmma_gemm(
    const _Float16* __restrict__ Aq,   // [T,K] quantized x (integer-valued f16)
    const _Float16* __restrict__ Bq,   // [N,K] quantized weight
    const float*    __restrict__ sxT,  // [K/32][T]  input dequant factors
    const float*    __restrict__ sw,   // [K/32][N/32] weight dequant factors
    const float*    __restrict__ bias, // [N]
    float*          __restrict__ out,  // [T,N]
    int T, int N, int K) {
    __shared__ _Float16 ldsA[2][128][LDP];
    __shared__ _Float16 ldsB[2][128][LDP];

    const int tid  = threadIdx.x;
    const int lane = tid & 31;
    const int wid  = tid >> 5;
    const int wm   = wid & 3;          // wave M quarter (rows 32*wm)
    const int wn   = wid >> 2;         // wave N half   (cols 64*wn)
    const int l16  = lane & 15;
    const int hig  = lane >> 4;        // lane half (0/1)
    const int k0   = hig * 8;          // A-fragment K base for this half

    const int m_base = blockIdx.y * 128;
    const int n_base = blockIdx.x * 128;
    const int KB  = K >> 5;
    const int NB  = N >> 5;

    // staging: 2 threads per row, each copies 16 halves (2 x b128)
    const int srow  = tid >> 1;
    const int shalf = (tid & 1) * 16;         // half-offset within row

    const _Float16* gA0 = &Aq[(size_t)(m_base + srow) * K + shalf];
    const _Float16* gB0 = &Bq[(size_t)(n_base + srow) * K + shalf];

    v8f acc[2][4] = {};
    const v8f zero = {};

    // prologue: prefetch k-blocks 0 and 1
#pragma unroll
    for (int p = 0; p < 2; ++p) {
        const _Float16* ga = gA0 + (size_t)p * 32;
        const _Float16* gb = gB0 + (size_t)p * 32;
        _Float16* la = &ldsA[p][srow][shalf];
        _Float16* lb = &ldsB[p][srow][shalf];
        lm_copy16(ga, la);     lm_copy16(ga + 8, la + 8);
        lm_copy16(gb, lb);     lm_copy16(gb + 8, lb + 8);
    }

    for (int kb = 0; kb < KB; ++kb) {
        const int cur = kb & 1;
        // current buffer's 4 async loads done; next buffer may stay in flight
        if (kb + 1 < KB) { LM_WAIT_ASYNC(4); } else { LM_WAIT_ASYNC(0); }
        __syncthreads();

        // A fragments: lane row = l16; K = {k0..k0+7, k0+16..k0+23}
        v16h af[2];
#pragma unroll
        for (int i = 0; i < 2; ++i) {
            const int row = wm * 32 + i * 16 + l16;
            v8h lo = *(const v8h*)&ldsA[cur][row][k0];
            v8h hi = *(const v8h*)&ldsA[cur][row][k0 + 16];
            af[i] = __builtin_shufflevector(lo, hi, 0, 1, 2, 3, 4, 5, 6, 7,
                                            8, 9, 10, 11, 12, 13, 14, 15);
        }
        // B fragments: lane col = l16; K = {k0..k0+7, k0+16..k0+23}
        v16h bf[4];
#pragma unroll
        for (int j = 0; j < 4; ++j) {
            const int row = wn * 64 + j * 16 + l16;
            v8h lo = *(const v8h*)&ldsB[cur][row][k0];
            v8h hi = *(const v8h*)&ldsB[cur][row][k0 + 16];
            bf[j] = __builtin_shufflevector(lo, hi, 0, 1, 2, 3, 4, 5, 6, 7,
                                            8, 9, 10, 11, 12, 13, 14, 15);
        }

        // per-row input scales: 8 consecutive floats thanks to [KB][T] layout
        v8f sx[2];
#pragma unroll
        for (int i = 0; i < 2; ++i)
            sx[i] = *(const v8f*)&sxT[(size_t)kb * T + m_base + wm * 32 +
                                      i * 16 + hig * 8];
        // per-32-N-block weight scales (wave-uniform)
        float swv[2];
#pragma unroll
        for (int h = 0; h < 2; ++h)
            swv[h] = sw[(size_t)kb * NB + (n_base >> 5) + wn * 2 + h];

        // 8 WMMAs, depth-1 software pipeline: scale tile t-1 while tile t runs
        v8f pcur = __builtin_amdgcn_wmma_f32_16x16x32_f16(
            false, af[0], false, bf[0], (short)0, zero, false, false);
#pragma unroll
        for (int t = 1; t < 8; ++t) {
            const int i = t >> 2, j = t & 3;
            v8f pn = __builtin_amdgcn_wmma_f32_16x16x32_f16(
                false, af[i], false, bf[j], (short)0, zero, false, false);
            const int pi = (t - 1) >> 2, pj = (t - 1) & 3;
            const float s = swv[pj >> 1];
#pragma unroll
            for (int r = 0; r < 8; ++r)
                acc[pi][pj][r] += pcur[r] * sx[pi][r] * s;
            pcur = pn;
        }
        {
            const float s = swv[1];
#pragma unroll
            for (int r = 0; r < 8; ++r)
                acc[1][3][r] += pcur[r] * sx[1][r] * s;
        }

        __syncthreads();   // everyone done reading buffer `cur`
        if (kb + 2 < KB) { // refill it for k-block kb+2
            const _Float16* ga = gA0 + (size_t)(kb + 2) * 32;
            const _Float16* gb = gB0 + (size_t)(kb + 2) * 32;
            _Float16* la = &ldsA[cur][srow][shalf];
            _Float16* lb = &ldsB[cur][srow][shalf];
            lm_copy16(ga, la);     lm_copy16(ga + 8, la + 8);
            lm_copy16(gb, lb);     lm_copy16(gb + 8, lb + 8);
        }
    }

    // epilogue: D lane holds column n, VGPR r holds row (r + 8*hig)
#pragma unroll
    for (int j = 0; j < 4; ++j) {
        const int n = n_base + wn * 64 + j * 16 + l16;
        const float bv = bias[n];
#pragma unroll
        for (int i = 0; i < 2; ++i) {
            const int trow = m_base + wm * 32 + i * 16 + hig * 8;
#pragma unroll
            for (int r = 0; r < 8; ++r)
                out[(size_t)(trow + r) * N + n] = acc[i][j][r] + bv;
        }
    }
}

// ---------------------------------------------------------------------------
extern "C" void kernel_launch(void* const* d_in, const int* in_sizes, int n_in,
                              void* d_out, int out_size, void* d_ws, size_t ws_size,
                              hipStream_t stream) {
    const float* x    = (const float*)d_in[0];   // [T,K]
    const float* w    = (const float*)d_in[1];   // [N,K]
    const float* bias = (const float*)d_in[2];   // [N]
    float* out = (float*)d_out;

    const int N = in_sizes[2];
    const int K = (int)((long long)in_sizes[1] / N);
    const int T = (int)((long long)in_sizes[0] / K);

    char* ws = (char*)d_ws;
    size_t off = 0;
    _Float16* xq = (_Float16*)(ws + off);
    off += (size_t)T * K * sizeof(_Float16); off = (off + 255) & ~(size_t)255;
    _Float16* wq = (_Float16*)(ws + off);
    off += (size_t)N * K * sizeof(_Float16); off = (off + 255) & ~(size_t)255;
    float* sxT = (float*)(ws + off);
    off += (size_t)(K / 32) * T * sizeof(float); off = (off + 255) & ~(size_t)255;
    float* sw = (float*)(ws + off);

    // quantize inputs
    const unsigned qx_blocks = (unsigned)((size_t)T * K / 256);   // wave per 32-block
    lm_quant_x<<<qx_blocks, 256, 0, stream>>>(x, xq, sxT, T, K);
    const unsigned qw_blocks = (unsigned)((N / 32) * (K / 32));
    lm_quant_w<<<qw_blocks, 256, 0, stream>>>(w, wq, sw, N, K);

    // WMMA GEMM
    dim3 grid(N / 128, T / 128);
    lm_wmma_gemm<<<grid, 256, 0, stream>>>(xq, wq, sxT, sw, bias, out, T, N, K);
}
